// GNNembedding_34136400069092
// MI455X (gfx1250) — compile-verified, matrix operand order checked by reading, not run on hardware
//
#include <hip/hip_runtime.h>

typedef __attribute__((ext_vector_type(16))) _Float16 v16h;
typedef __attribute__((ext_vector_type(8)))  _Float16 v8h;
typedef __attribute__((ext_vector_type(8)))  float    v8f;

#define N_NODES 100000
#define N_EDGES 1200000
#define DIM 64
#define KTOT 128                 // combined K: [mean | x]
#define LDS_STRIDE 136           // 128 + 8 halves pad -> 272B rows, conflict-free b128
#define M_TILES (N_NODES / 16)   // 6250 exactly

// ---------------------------------------------------------------------------
// Zero the agg+cnt region (graph-capture-safe memset).
// ---------------------------------------------------------------------------
__global__ void zero_f32(float* __restrict__ p, int n) {
  int i = blockIdx.x * blockDim.x + threadIdx.x;
  if (i < n) p[i] = 0.0f;
}

// ---------------------------------------------------------------------------
// One-time weight prep: Wt[n][k] = f16( k<64 ? Wn[k][n] : Ws[k-64][n] )
// i.e. transposed+converted combined weight, row-major [64][128] f16 (16 KB).
// Transposed layout makes each lane's per-step B fragment two contiguous
// 16-byte chunks (ds_load_b128 from LDS later).
// ---------------------------------------------------------------------------
__global__ void prep_weights(const float* __restrict__ Wn,
                             const float* __restrict__ Ws,
                             _Float16* __restrict__ Wt) {
  int i = blockIdx.x * blockDim.x + threadIdx.x;  // 64*128 = 8192
  if (i >= DIM * KTOT) return;
  int n = i >> 7;          // output column 0..63
  int k = i & 127;         // combined input dim 0..127
  float v = (k < DIM) ? Wn[k * DIM + n] : Ws[(k - DIM) * DIM + n];
  Wt[i] = (_Float16)v;
}

// ---------------------------------------------------------------------------
// Edge scatter: agg[dst] += feat[src], cnt[dst] += 1.
// 16 threads/edge, float4 per thread (256B coalesced gather per edge).
// f32 atomics resolve in L2: the 25.6MB accumulator fits in the 192MB L2.
// ---------------------------------------------------------------------------
__global__ __launch_bounds__(256) void scatter_mean_accum(
    const float* __restrict__ feat,
    const int*   __restrict__ src,
    const int*   __restrict__ dst,
    float*       __restrict__ agg,
    float*       __restrict__ cnt) {
  int gid = blockIdx.x * blockDim.x + threadIdx.x;
  int e   = gid >> 4;
  int d4  = (gid & 15) * 4;
  if (e >= N_EDGES) return;
  int s = src[e];
  int t = dst[e];
  const float4 v = *(const float4*)(feat + (size_t)s * DIM + d4);
  float* a = agg + (size_t)t * DIM + d4;
  atomicAdd(a + 0, v.x);
  atomicAdd(a + 1, v.y);
  atomicAdd(a + 2, v.z);
  atomicAdd(a + 3, v.w);
  if (d4 == 0) atomicAdd(cnt + t, 1.0f);
}

// ---------------------------------------------------------------------------
// Fused SAGEConv GEMM: out = relu([mean | x] @ Wt^T + bias)
// One wave per 16-row strip (16x64 output): 4 K-steps x 4 N-tiles = 16 WMMAs.
// A fragment loaded once per K-step and reused across all 4 N-tiles (4x less
// HBM traffic for the streamed features). Weights staged in LDS (16 KB,
// padded rows -> bank-conflict-free ds_load_b128 B fragments).
//
// Fragment layouts (CDNA5 ISA 7.12.2, wave32):
//   A 16x32 f16 : row = lane&15, khalf = (lane>>4)*8;
//                 VGPR j<4 -> K=khalf+2j ; j>=4 -> K=16+khalf+2(j-4)
//                 => elements 0..7 = K khalf..khalf+7 (contiguous),
//                    elements 8..15 = K 16+khalf..16+khalf+7 (contiguous)
//   B 32x16 f16 : col = lane&15, same K packing (symmetric)
//   C/D 16x16 f32: VGPR r -> row r + 8*(lane>>4), col = lane&15
// ---------------------------------------------------------------------------
__global__ __launch_bounds__(256) void sage_wmma(
    const float*    __restrict__ xf,    // self features  [N,64]
    const float*    __restrict__ agg,   // summed neighbor features [N,64]
    const float*    __restrict__ cnt,   // degree [N]
    const _Float16* __restrict__ Wt,    // transposed combined weights [64][128]
    const float*    __restrict__ bias,  // [64]
    float*          __restrict__ out)   // [N,64]
{
  __shared__ _Float16 wlds[DIM * LDS_STRIDE];  // 17408 B

  const int tid = threadIdx.x;

  // ---- Cooperative LDS fill: 8192 halves, 16B chunks, padded rows ----
#pragma unroll
  for (int cchunk = 0; cchunk < 4; ++cchunk) {
    int i8  = tid + cchunk * 256;        // chunk index 0..1023
    int idx = i8 * 8;                    // half index into Wt
    int n   = idx >> 7;                  // 0..63
    int k   = idx & 127;                 // multiple of 8
    *(v8h*)(&wlds[n * LDS_STRIDE + k]) = *(const v8h*)(Wt + idx);
  }
  __syncthreads();

  const int lane = tid & 31;
  const int mt   = blockIdx.x * 8 + (tid >> 5);  // one wave per 16-row strip
  if (mt >= M_TILES) return;                     // wave-uniform (after barrier)

  const int row   = mt * 16 + (lane & 15);
  const int khalf = (lane >> 4) * 8;
  const float invc = 1.0f / fmaxf(cnt[row], 1.0f);

  v8f c[4];
#pragma unroll
  for (int nt = 0; nt < 4; ++nt)
    c[nt] = (v8f){0.f, 0.f, 0.f, 0.f, 0.f, 0.f, 0.f, 0.f};

#pragma unroll
  for (int s = 0; s < 4; ++s) {
    // ---- A fragment: two contiguous 8-float runs, global_load_b128 x4 ----
    const bool  isMean = (s < 2);
    const float* srcp  = isMean ? (agg + (size_t)row * DIM)
                                : (xf  + (size_t)row * DIM);
    const int   k0     = isMean ? (s * 32) : (s * 32 - DIM);
    const float sc     = isMean ? invc : 1.0f;
    const float4 f0 = *(const float4*)(srcp + k0 + khalf);
    const float4 f1 = *(const float4*)(srcp + k0 + khalf + 4);
    const float4 f2 = *(const float4*)(srcp + k0 + 16 + khalf);
    const float4 f3 = *(const float4*)(srcp + k0 + 16 + khalf + 4);
    v16h a;
    a[0]  = (_Float16)(f0.x * sc); a[1]  = (_Float16)(f0.y * sc);
    a[2]  = (_Float16)(f0.z * sc); a[3]  = (_Float16)(f0.w * sc);
    a[4]  = (_Float16)(f1.x * sc); a[5]  = (_Float16)(f1.y * sc);
    a[6]  = (_Float16)(f1.z * sc); a[7]  = (_Float16)(f1.w * sc);
    a[8]  = (_Float16)(f2.x * sc); a[9]  = (_Float16)(f2.y * sc);
    a[10] = (_Float16)(f2.z * sc); a[11] = (_Float16)(f2.w * sc);
    a[12] = (_Float16)(f3.x * sc); a[13] = (_Float16)(f3.y * sc);
    a[14] = (_Float16)(f3.z * sc); a[15] = (_Float16)(f3.w * sc);

    // ---- 4 N-tiles reuse this A fragment ----
#pragma unroll
    for (int nt = 0; nt < 4; ++nt) {
      const int coln = nt * 16 + (lane & 15);
      const v8h b0 = *(const v8h*)(&wlds[coln * LDS_STRIDE + s * 32 + khalf]);
      const v8h b1 = *(const v8h*)(&wlds[coln * LDS_STRIDE + s * 32 + 16 + khalf]);
      const v16h b = __builtin_shufflevector(b0, b1, 0, 1, 2, 3, 4, 5, 6, 7,
                                             8, 9, 10, 11, 12, 13, 14, 15);
      c[nt] = __builtin_amdgcn_wmma_f32_16x16x32_f16(false, a, false, b,
                                                     (short)0, c[nt],
                                                     false, false);
    }
  }

  // ---- Epilogue: bias + ReLU, 16 lanes cover a 64B row segment per store ----
#pragma unroll
  for (int nt = 0; nt < 4; ++nt) {
    const int coln = nt * 16 + (lane & 15);
    const float bv = bias[coln];
#pragma unroll
    for (int r = 0; r < 8; ++r) {
      const int m = mt * 16 + (lane >> 4) * 8 + r;
      out[(size_t)m * DIM + coln] = fmaxf(c[nt][r] + bv, 0.0f);
    }
  }
}

// ---------------------------------------------------------------------------
// Host launcher. Inputs (setup_inputs order):
//  0 x[N,64] f32, 1 edge_index[2,E] int, 2 W_neigh1, 3 W_self1, 4 b1,
//  5 W_neigh2, 6 W_self2, 7 b2.  Output: h2 [N,64] f32.
// Workspace: agg (N*64 f32) | cnt (N f32) | h1 (N*64 f32) | Wt1,Wt2 (f16).
// ---------------------------------------------------------------------------
extern "C" void kernel_launch(void* const* d_in, const int* in_sizes, int n_in,
                              void* d_out, int out_size, void* d_ws, size_t ws_size,
                              hipStream_t stream) {
  const float* x   = (const float*)d_in[0];
  const int*   ei  = (const int*)d_in[1];
  const float* Wn1 = (const float*)d_in[2];
  const float* Ws1 = (const float*)d_in[3];
  const float* b1  = (const float*)d_in[4];
  const float* Wn2 = (const float*)d_in[5];
  const float* Ws2 = (const float*)d_in[6];
  const float* b2  = (const float*)d_in[7];
  float* out = (float*)d_out;

  float*    agg = (float*)d_ws;                   // N*64 floats
  float*    cnt = agg + (size_t)N_NODES * DIM;    // N floats
  float*    h1  = cnt + N_NODES;                  // N*64 floats
  _Float16* wt1 = (_Float16*)(h1 + (size_t)N_NODES * DIM);  // 8192 halves
  _Float16* wt2 = wt1 + DIM * KTOT;                          // 8192 halves

  const int* src = ei;             // edge_index[0,:]
  const int* dst = ei + N_EDGES;   // edge_index[1,:]

  const int zn      = N_NODES * (DIM + 1);
  const int zblocks = (zn + 255) / 256;
  const int sblocks = (N_EDGES * 16 + 255) / 256;
  const int gblocks = (M_TILES + 7) / 8;          // 8 waves (strips) per block
  const int pblocks = (DIM * KTOT + 255) / 256;

  // ---- Weight prep (both layers up front) ----
  prep_weights<<<pblocks, 256, 0, stream>>>(Wn1, Ws1, wt1);
  prep_weights<<<pblocks, 256, 0, stream>>>(Wn2, Ws2, wt2);

  // ---- Layer 1 ----
  zero_f32<<<zblocks, 256, 0, stream>>>(agg, zn);
  scatter_mean_accum<<<sblocks, 256, 0, stream>>>(x, src, dst, agg, cnt);
  sage_wmma<<<gblocks, 256, 0, stream>>>(x, agg, cnt, wt1, b1, h1);

  // ---- Layer 2 ----
  zero_f32<<<zblocks, 256, 0, stream>>>(agg, zn);
  scatter_mean_accum<<<sblocks, 256, 0, stream>>>(h1, src, dst, agg, cnt);
  sage_wmma<<<gblocks, 256, 0, stream>>>(h1, agg, cnt, wt2, b2, out);
}
